// Attention_27814208209082
// MI455X (gfx1250) — matrix operand order repaired and checked by reading, not compiled, not run on hardware
//
#include <hip/hip_runtime.h>
#include <hip/hip_bf16.h>

// Problem constants
#define B_   4
#define S_   2048
#define D_   768
#define H_   12
#define HD_  64
#define M_   (B_*S_)    // 8192 rows
#define N3D_ (3*D_)     // 2304

#define NEG_INF_F (-1000000000.0f)

typedef __attribute__((ext_vector_type(16))) __bf16 v16bf;
typedef __attribute__((ext_vector_type(8)))  __bf16 v8bf;
typedef __attribute__((ext_vector_type(8)))  float  v8f;
typedef __attribute__((ext_vector_type(4)))  float  v4f;
typedef __attribute__((ext_vector_type(4)))  unsigned int v4u;
typedef __attribute__((ext_vector_type(4)))  int    v4i;
typedef __attribute__((ext_vector_type(8)))  int    v8i;

union Frag { v16bf v; v8bf h[2]; };

#if defined(__has_builtin)
#if __has_builtin(__builtin_amdgcn_tensor_load_to_lds) && __has_builtin(__builtin_amdgcn_s_wait_tensorcnt)
#define HAVE_TDM 1
#endif
#endif

__device__ __forceinline__ v8f wmma_bf16(const Frag& a, const Frag& b, v8f c) {
  // D = A(16x32 bf16) * B(32x16 bf16) + C(16x16 f32)
  return __builtin_amdgcn_wmma_f32_16x16x32_bf16(
      /*neg_a=*/false, a.v, /*neg_b=*/false, b.v,
      /*c_mod=*/(short)0, c, /*reuse_a=*/false, /*reuse_b=*/false);
}

#if HAVE_TDM
// LDS byte-offset of a __shared__ object (generic -> addrspace(3) cast).
__device__ __forceinline__ unsigned int lds_off(const void* p) {
  return (unsigned int)(unsigned long long)(__attribute__((address_space(3))) const char*)p;
}

// Issue one TDM 2D tile load: tile_d0 x tile_d1 bf16 elements from a
// row-major [tensor_d1][tensor_d0-strided] tensor into LDS, padding each
// 16-DWORD row with 4 DWORDs (=> LDS row stride 40 bf16 elems, bank-safe).
// D# layout per CDNA5 ISA 8.3/8.4. Tracked by TENSORcnt.
__device__ __forceinline__ void tdm_load_2d(unsigned int ldsAddr, const __bf16* gptr,
                                            unsigned int tensor_d0, unsigned int tensor_d1,
                                            unsigned int tile_d0, unsigned int tile_d1,
                                            unsigned int stride0) {
  unsigned long long ga = (unsigned long long)gptr;
  v4u g0;
  g0.x = 1u;                                             // count=1 (valid user D#)
  g0.y = ldsAddr;                                        // lds_addr [63:32]
  g0.z = (unsigned int)ga;                               // global_addr [95:64]
  g0.w = ((unsigned int)(ga >> 32) & 0x1FFFFFFu) | (2u << 30);  // addr[56:32] | type=2
  v8i g1;
  g1[0] = (1 << 16)        // data_size = 2 bytes
        | (1 << 20)        // pad_enable
        | (3 << 22)        // pad_interval: 16 DWORDs
        | (3 << 25);       // pad_amount: 4 DWORDs
  g1[1] = (int)((tensor_d0 & 0xFFFFu) << 16);                       // dim0[15:0] @ bits63:48
  g1[2] = (int)((tensor_d0 >> 16) | ((tensor_d1 & 0xFFFFu) << 16)); // dim0[31:16], dim1[15:0]
  g1[3] = (int)((tensor_d1 >> 16) | (tile_d0 << 16));               // dim1[31:16], tile_dim0
  g1[4] = (int)tile_d1;                                             // tile_dim1 (tile_dim2=0)
  g1[5] = (int)stride0;                                             // dim0_stride[31:0]
  g1[6] = 0;
  g1[7] = 0;
  v4i z4 = {0, 0, 0, 0};
#if __clang_major__ >= 23
  v8i z8 = {0, 0, 0, 0, 0, 0, 0, 0};
  __builtin_amdgcn_tensor_load_to_lds(g0, g1, z4, z4, z8, 0);
#else
  __builtin_amdgcn_tensor_load_to_lds(g0, g1, z4, z4, 0);
#endif
}
#endif  // HAVE_TDM

// ---------------------------------------------------------------------------
// Pass 0a: f32 -> bf16 elementwise (x). total must be a multiple of 8.
// ---------------------------------------------------------------------------
__global__ void cvt_f32_bf16(const float* __restrict__ in,
                             __bf16* __restrict__ out, int total) {
  int i = (blockIdx.x * blockDim.x + threadIdx.x) * 8;
  if (i >= total) return;
  v4f a = *(const v4f*)(in + i);
  v4f b = *(const v4f*)(in + i + 4);
  v8bf o;
#pragma unroll
  for (int j = 0; j < 4; ++j) { o[j] = (__bf16)a[j]; o[4 + j] = (__bf16)b[j]; }
  *(v8bf*)(out + i) = o;
}

// ---------------------------------------------------------------------------
// Pass 0b: transpose-convert weights: in[R][C] f32 -> out[C][R] bf16
// ---------------------------------------------------------------------------
__global__ void transpose_f32_bf16(const float* __restrict__ in,
                                   __bf16* __restrict__ out, int R, int C) {
  int i = blockIdx.x * blockDim.x + threadIdx.x;
  if (i >= R * C) return;
  int c = i / R, r = i - c * R;
  out[i] = (__bf16)in[(size_t)r * C + c];
}

// ---------------------------------------------------------------------------
// Shared per-k-step WMMA compute on a staged 128x32 A tile and 128x32 Bt tile.
// A-frag: lane r=lane&15 row, K chunks at (lane>>4)*8 / +16 (ISA 7.12.2).
// B-frag: lane = column, K chunks at (lane>>4)*8 / +16 (same as A).
// ---------------------------------------------------------------------------
__device__ __forceinline__ void compute_tile(const __bf16 (*__restrict__ As)[40],
                                             const __bf16 (*__restrict__ Bs)[40],
                                             int waveM, int waveN, int r, int hi,
                                             v8f acc[2][4]) {
  Frag a[2], b[4];
#pragma unroll
  for (int mt = 0; mt < 2; ++mt) {
    int rr = waveM * 32 + mt * 16 + r;
    a[mt].h[0] = *(const v8bf*)&As[rr][hi * 8];
    a[mt].h[1] = *(const v8bf*)&As[rr][hi * 8 + 16];
  }
#pragma unroll
  for (int nt = 0; nt < 4; ++nt) {
    int cc = waveN * 64 + nt * 16 + r;
    b[nt].h[0] = *(const v8bf*)&Bs[cc][hi * 8];
    b[nt].h[1] = *(const v8bf*)&Bs[cc][hi * 8 + 16];
  }
#pragma unroll
  for (int mt = 0; mt < 2; ++mt)
#pragma unroll
    for (int nt = 0; nt < 4; ++nt)
      acc[mt][nt] = wmma_bf16(a[mt], b[nt], acc[mt][nt]);
}

// ---------------------------------------------------------------------------
// Main-loop macro: TDM double-buffered staging when available, otherwise
// register staging via global_load_b128 + ds_store_b128.
// A: [M,768] bf16 row-major, Bt: [N,768] bf16 N-major; acc produced per wave.
// ---------------------------------------------------------------------------
#if HAVE_TDM
#define GEMM_MAIN_LOOP(Aptr, Btptr, rowsA, rowsB)                                   \
  __shared__ __align__(32) __bf16 As[2][128][40];                                   \
  __shared__ __align__(32) __bf16 Bs[2][128][40];                                   \
  v8f acc[2][4] = {};                                                               \
  const int NK = D_ / 32;                                                           \
  if (wid == 0) {                                                                   \
    tdm_load_2d(lds_off(&As[0][0][0]), (Aptr) + (size_t)mBase * D_, D_, (rowsA),    \
                32, 128, D_);                                                       \
    tdm_load_2d(lds_off(&Bs[0][0][0]), (Btptr) + (size_t)nBase * D_, D_, (rowsB),   \
                32, 128, D_);                                                       \
  }                                                                                 \
  for (int kt = 0; kt < NK; ++kt) {                                                 \
    int cur = kt & 1, nxt = cur ^ 1;                                                \
    if (wid == 0) {                                                                 \
      if (kt + 1 < NK) {                                                            \
        tdm_load_2d(lds_off(&As[nxt][0][0]),                                        \
                    (Aptr) + (size_t)mBase * D_ + (kt + 1) * 32, D_, (rowsA),       \
                    32, 128, D_);                                                   \
        tdm_load_2d(lds_off(&Bs[nxt][0][0]),                                        \
                    (Btptr) + (size_t)nBase * D_ + (kt + 1) * 32, D_, (rowsB),      \
                    32, 128, D_);                                                   \
        __builtin_amdgcn_s_wait_tensorcnt(2);                                       \
      } else {                                                                      \
        __builtin_amdgcn_s_wait_tensorcnt(0);                                       \
      }                                                                             \
    }                                                                               \
    __syncthreads();                                                                \
    compute_tile(As[cur], Bs[cur], waveM, waveN, r, hi, acc);                       \
    __syncthreads();                                                                \
  }
#else
#define GEMM_MAIN_LOOP(Aptr, Btptr, rowsA, rowsB)                                   \
  __shared__ __align__(32) __bf16 As[128][40];                                      \
  __shared__ __align__(32) __bf16 Bs[128][40];                                      \
  v8f acc[2][4] = {};                                                               \
  const int row  = threadIdx.x >> 1;                                                \
  const int half = (threadIdx.x & 1) * 16;                                          \
  const __bf16* aSrc = (Aptr) + (size_t)(mBase + row) * D_ + half;                  \
  const __bf16* bSrc = (Btptr) + (size_t)(nBase + row) * D_ + half;                 \
  for (int kt = 0; kt < D_ / 32; ++kt) {                                            \
    Frag t0, t1;                                                                    \
    t0.v = *(const v16bf*)(aSrc + kt * 32);                                         \
    t1.v = *(const v16bf*)(bSrc + kt * 32);                                         \
    if (kt + 1 < D_ / 32) {                                                         \
      __builtin_prefetch(aSrc + (kt + 1) * 32, 0, 1);                               \
      __builtin_prefetch(bSrc + (kt + 1) * 32, 0, 1);                               \
    }                                                                               \
    *(v8bf*)&As[row][half]     = t0.h[0];                                           \
    *(v8bf*)&As[row][half + 8] = t0.h[1];                                           \
    *(v8bf*)&Bs[row][half]     = t1.h[0];                                           \
    *(v8bf*)&Bs[row][half + 8] = t1.h[1];                                           \
    __syncthreads();                                                                \
    compute_tile(As, Bs, waveM, waveN, r, hi, acc);                                 \
    __syncthreads();                                                                \
  }
#endif

// ---------------------------------------------------------------------------
// QKV GEMM:  A[M_,D_] x Bt[N3D_,D_] + bias. Block tile 128x128, K-step 32,
// 8 waves (4x2), each wave 2x4 WMMA tiles. Epilogue scatters into Q,K
// [B,H,S,HD] and V transposed [B,H,HD,S] (all bf16).
// ---------------------------------------------------------------------------
__global__ __launch_bounds__(256) void qkv_gemm(
    const __bf16* __restrict__ A, const __bf16* __restrict__ Bt,
    const float* __restrict__ bias,
    __bf16* __restrict__ Qo, __bf16* __restrict__ Ko, __bf16* __restrict__ Vt) {
  const int nBase = blockIdx.x * 128;
  const int mBase = blockIdx.y * 128;
  const int lane = threadIdx.x & 31;
  const int wid  = threadIdx.x >> 5;
  const int waveM = wid >> 1;     // 0..3 -> 32 rows
  const int waveN = wid & 1;      // 0..1 -> 64 cols
  const int r  = lane & 15;
  const int hi = lane >> 4;

  GEMM_MAIN_LOOP(A, Bt, M_, N3D_)

  // Epilogue: C layout row = hi*8+v, col = r (per 16x16 tile)
#pragma unroll
  for (int mt = 0; mt < 2; ++mt) {
    int rowB = mBase + waveM * 32 + mt * 16 + hi * 8;
#pragma unroll
    for (int nt = 0; nt < 4; ++nt) {
      int n = nBase + waveN * 64 + nt * 16 + r;
      int which = n / D_;
      int f = n - which * D_;
      int h = f >> 6, hd = f & 63;
      float bn = bias[n];
#pragma unroll
      for (int v = 0; v < 8; ++v) {
        int m = rowB + v;
        int bb = m >> 11, s = m & (S_ - 1);
        __bf16 bv = (__bf16)(acc[mt][nt][v] + bn);
        size_t bh = (size_t)bb * H_ + h;
        if (which == 0)      Qo[(bh * S_ + s) * HD_ + hd] = bv;
        else if (which == 1) Ko[(bh * S_ + s) * HD_ + hd] = bv;
        else                 Vt[(bh * HD_ + hd) * S_ + s] = bv;  // V transposed
      }
    }
  }
}

// ---------------------------------------------------------------------------
// Fused causal attention. Grid: (S/64, B*H). 128 threads = 4 waves.
// Wave w owns 16 query rows m0 = qb*64 + w*16. Streams 32-key blocks.
// Faithful masking: t = masked ? NEG_INF : s*(1-NEG_INF) + NEG_INF.
// ---------------------------------------------------------------------------
__global__ __launch_bounds__(128) void attn_fused(
    const __bf16* __restrict__ Q, const __bf16* __restrict__ K,
    const __bf16* __restrict__ Vt, __bf16* __restrict__ ctx) {
  const int qb = blockIdx.x;
  const int bh = blockIdx.y;
  const int bb = bh / H_, h = bh - bb * H_;
  const int lane = threadIdx.x & 31;
  const int w = threadIdx.x >> 5;
  const int r  = lane & 15;
  const int hi = lane >> 4;
  const int m0 = qb * 64 + w * 16;

  const __bf16* Qb = Q  + (size_t)bh * S_ * HD_;
  const __bf16* Kb = K  + (size_t)bh * S_ * HD_;
  const __bf16* Vb = Vt + (size_t)bh * HD_ * S_;

  __shared__ __align__(32) __bf16 Ps[4][16][40];  // per-wave P transpose pad

  // Q A-fragments (held for whole kernel): 16 rows x 64 (two K=32 frags)
  Frag qa[2];
#pragma unroll
  for (int f = 0; f < 2; ++f) {
    const __bf16* p = Qb + (size_t)(m0 + r) * HD_ + f * 32 + hi * 16;
    qa[f].h[0] = *(const v8bf*)p;
    qa[f].h[1] = *(const v8bf*)(p + 8);
  }

  v8f o[4] = {};
  float mrun[8], lrun[8];
#pragma unroll
  for (int v = 0; v < 8; ++v) { mrun[v] = -3.0e38f; lrun[v] = 0.0f; }
  const int rowBase = m0 + hi * 8;
  const float CMUL = 1.0f - NEG_INF_F;       // ~1e9+1
  const int nkb = ((m0 + 15) >> 5) + 1;      // causal key-block count

  for (int kb = 0; kb < nkb; ++kb) {
    const int k0 = kb * 32;
    float tb[2][8];
#pragma unroll
    for (int kt = 0; kt < 2; ++kt) {
      const int key = k0 + kt * 16 + r;      // this lane's key column
      Frag bk0, bk1;
      bk0.v = *(const v16bf*)(Kb + (size_t)key * HD_ +      hi * 16);
      bk1.v = *(const v16bf*)(Kb + (size_t)key * HD_ + 32 + hi * 16);
      v8f sc = {};
      sc = wmma_bf16(qa[0], bk0, sc);
      sc = wmma_bf16(qa[1], bk1, sc);
      const int col = k0 + kt * 16 + r;
#pragma unroll
      for (int v = 0; v < 8; ++v) {
        int rowq = rowBase + v;
        float ss = sc[v] * 0.125f;           // 1/sqrt(64)
        tb[kt][v] = (col > rowq) ? NEG_INF_F : fmaf(ss, CMUL, NEG_INF_F);
      }
    }
    // Online softmax: row reductions across the 16 lanes holding each row
    float fac[8];
#pragma unroll
    for (int v = 0; v < 8; ++v) {
      float mx = fmaxf(tb[0][v], tb[1][v]);
#pragma unroll
      for (int off = 8; off >= 1; off >>= 1) mx = fmaxf(mx, __shfl_xor(mx, off, 32));
      float mnew = fmaxf(mrun[v], mx);
      fac[v] = __expf(mrun[v] - mnew);
      float p0 = __expf(tb[0][v] - mnew);
      float p1 = __expf(tb[1][v] - mnew);
      tb[0][v] = p0; tb[1][v] = p1;
      float ps = p0 + p1;
#pragma unroll
      for (int off = 8; off >= 1; off >>= 1) ps += __shfl_xor(ps, off, 32);
      lrun[v] = lrun[v] * fac[v] + ps;
      mrun[v] = mnew;
    }
#pragma unroll
    for (int ct = 0; ct < 4; ++ct)
#pragma unroll
      for (int v = 0; v < 8; ++v) o[ct][v] *= fac[v];

    // P: C-layout -> A-layout via per-wave LDS pad
#pragma unroll
    for (int kt = 0; kt < 2; ++kt)
#pragma unroll
      for (int v = 0; v < 8; ++v)
        Ps[w][hi * 8 + v][kt * 16 + r] = (__bf16)tb[kt][v];
    asm volatile("s_wait_dscnt 0" ::: "memory");
    Frag pa;
    pa.h[0] = *(const v8bf*)&Ps[w][r][hi * 16];
    pa.h[1] = *(const v8bf*)&Ps[w][r][hi * 16 + 8];

    // o += P(16x32) * V(32x64) using transposed-V B-fragments
#pragma unroll
    for (int ct = 0; ct < 4; ++ct) {
      Frag bv;
      int hd = ct * 16 + r;
      bv.v = *(const v16bf*)(Vb + (size_t)hd * S_ + k0 + hi * 16);
      o[ct] = wmma_bf16(pa, bv, o[ct]);
    }
  }

  // Normalize and write ctx as bf16 [B,S,D]
#pragma unroll
  for (int ct = 0; ct < 4; ++ct) {
    int hd = ct * 16 + r;
#pragma unroll
    for (int v = 0; v < 8; ++v) {
      int s = rowBase + v;
      float val = o[ct][v] / lrun[v];
      ctx[((size_t)bb * S_ + s) * D_ + h * HD_ + hd] = (__bf16)val;
    }
  }
}

// ---------------------------------------------------------------------------
// Output projection: ctx[M_,D_] bf16 x Wo_t[D_,D_] bf16 + b_o -> f32 out
// ---------------------------------------------------------------------------
__global__ __launch_bounds__(256) void out_gemm(
    const __bf16* __restrict__ A, const __bf16* __restrict__ Bt,
    const float* __restrict__ bias, float* __restrict__ out) {
  const int nBase = blockIdx.x * 128;
  const int mBase = blockIdx.y * 128;
  const int lane = threadIdx.x & 31;
  const int wid  = threadIdx.x >> 5;
  const int waveM = wid >> 1;
  const int waveN = wid & 1;
  const int r  = lane & 15;
  const int hi = lane >> 4;

  GEMM_MAIN_LOOP(A, Bt, M_, D_)

#pragma unroll
  for (int mt = 0; mt < 2; ++mt) {
    int rowB = mBase + waveM * 32 + mt * 16 + hi * 8;
#pragma unroll
    for (int nt = 0; nt < 4; ++nt) {
      int n = nBase + waveN * 64 + nt * 16 + r;
      float bn = bias[n];
#pragma unroll
      for (int v = 0; v < 8; ++v)
        out[(size_t)(rowB + v) * D_ + n] = acc[mt][nt][v] + bn;
    }
  }
}

// ---------------------------------------------------------------------------
// Host-side orchestration (all on `stream`, graph-capture safe)
// ---------------------------------------------------------------------------
extern "C" void kernel_launch(void* const* d_in, const int* in_sizes, int n_in,
                              void* d_out, int out_size, void* d_ws, size_t ws_size,
                              hipStream_t stream) {
  const float* x     = (const float*)d_in[0];  // [B,S,D]
  const float* W_qkv = (const float*)d_in[1];  // [D,3D]
  const float* b_qkv = (const float*)d_in[2];  // [3D]
  const float* W_o   = (const float*)d_in[3];  // [D,D]
  const float* b_o   = (const float*)d_in[4];  // [D]
  float* out = (float*)d_out;

  char* ws = (char*)d_ws;
  const size_t SZ_XB   = (size_t)M_ * D_ * 2;       // 12.58 MB
  const size_t SZ_WQKV = (size_t)N3D_ * D_ * 2;     //  3.54 MB
  const size_t SZ_WO   = (size_t)D_ * D_ * 2;       //  1.18 MB
  const size_t SZ_HEAD = (size_t)B_ * H_ * S_ * HD_ * 2;
  size_t off = 0;
  __bf16* xb    = (__bf16*)(ws + off); off += SZ_XB;
  __bf16* wqkvt = (__bf16*)(ws + off); off += SZ_WQKV;
  __bf16* wot   = (__bf16*)(ws + off); off += SZ_WO;
  __bf16* Qb    = (__bf16*)(ws + off); off += SZ_HEAD;
  __bf16* Kb    = (__bf16*)(ws + off); off += SZ_HEAD;
  __bf16* Vtb   = (__bf16*)(ws + off); off += SZ_HEAD;   // [B,H,HD,S]
  __bf16* ctxb  = (__bf16*)(ws + off); off += SZ_XB;     // ~67.6 MB total

  // Pass 0: precision conversion + weight transpose
  cvt_f32_bf16<<<(M_ * D_) / (256 * 8), 256, 0, stream>>>(x, xb, M_ * D_);
  transpose_f32_bf16<<<(N3D_ * D_ + 255) / 256, 256, 0, stream>>>(W_qkv, wqkvt, D_, N3D_);
  transpose_f32_bf16<<<(D_ * D_ + 255) / 256, 256, 0, stream>>>(W_o, wot, D_, D_);

  // Pass 1: QKV projection (writes Q,K head-major; V transposed)
  qkv_gemm<<<dim3(N3D_ / 128, M_ / 128), 256, 0, stream>>>(xb, wqkvt, b_qkv, Qb, Kb, Vtb);

  // Pass 2: fused causal attention -> ctx bf16 [B,S,D]
  attn_fused<<<dim3(S_ / 64, B_ * H_), 128, 0, stream>>>(Qb, Kb, Vtb, ctxb);

  // Pass 3: output projection -> f32 d_out
  out_gemm<<<dim3(D_ / 128, M_ / 128), 256, 0, stream>>>(ctxb, wot, b_o, out);
}
